// MixerNetwork_9182640079047
// MI455X (gfx1250) — matrix operand order repaired and checked by reading, not compiled, not run on hardware
//
#include <hip/hip_runtime.h>
#include <math.h>

typedef _Float16 v16h __attribute__((ext_vector_type(16)));
typedef _Float16 v8h  __attribute__((ext_vector_type(8)));
typedef float    v8f  __attribute__((ext_vector_type(8)));

#define HP 128      // padded hidden width (true 100)
#define KT_H 4      // 128/32 K-chunks
#define NT_H 8      // 128/16 N-tiles

// ---------------------------------------------------------------- fast math

// Branch-free tanh on the HW transcendental pipe:
// tanh(x) = 1 - 2 / (exp(2x) + 1).  Safe at +-inf (exp->inf => 1, exp->0 => -1).
__device__ __forceinline__ float fast_tanh(float x) {
  const float t = __expf(2.0f * x);                 // v_exp_f32
  const float r = __builtin_amdgcn_rcpf(t + 1.0f);  // v_rcp_f32
  return fmaf(-2.0f, r, 1.0f);
}

__device__ __forceinline__ float fast_exp(float x) { return __expf(x); }

// ---------------------------------------------------------------- WMMA utils

__device__ __forceinline__ v8f wmma_f16(v16h a, v16h b, v8f c) {
  // D = A(16x32 f16) * B(32x16 f16) + C(16x16 f32)
  return __builtin_amdgcn_wmma_f32_16x16x32_f16(false, a, false, b, (short)0, c,
                                                false, false);
}

__device__ __forceinline__ v8f zero_v8f() {
  v8f z;
#pragma unroll
  for (int i = 0; i < 8; ++i) z[i] = 0.0f;
  return z;
}

// B tile (32x16) packed as 512 contiguous f16 per tile: lane*16 .. lane*16+15
__device__ __forceinline__ v16h load_btile(const _Float16* __restrict__ W,
                                           int tile, int lane) {
  return *(const v16h*)(W + (size_t)tile * 512 + lane * 16);
}

// A tile (16x32) from LDS activation buffer h[16][HP] (row-major f16).
// Doc layout: lanes 0-15 row m=lane, K = kt*32 + {0..7,16..23};
//             lanes 16-31 row m=lane-16, K = kt*32 + {8..15,24..31}.
__device__ __forceinline__ v16h load_atile_lds(const _Float16* h, int lane, int kt) {
  const int m  = lane & 15;
  const int kb = kt * 32 + ((lane >> 4) << 3);
  const _Float16* p = h + m * HP + kb;
  v8h lo = *(const v8h*)(p);
  v8h hi = *(const v8h*)(p + 16);
  v16h a;
#pragma unroll
  for (int j = 0; j < 8; ++j) { a[j] = lo[j]; a[j + 8] = hi[j]; }
  return a;
}

// ----------------------------------------------------------------- layers

// Layer 1: Gaussian basis (3 features, K=0..2) -> 128 (true 100), tanh, to LDS
__device__ __forceinline__ void layer_first(_Float16* h, float d,
    const _Float16* __restrict__ W1, const float* __restrict__ b1, int lane) {
  const float g0 = fast_exp(-0.5f * d * d);
  const float g1 = fast_exp(-0.5f * (d - 1.5f) * (d - 1.5f));
  const float g2 = fast_exp(-0.5f * (d - 3.0f) * (d - 3.0f));
  const bool lo = lane < 16;  // lanes>=16 hold K=8..15,24..31 -> zero padding
  const _Float16 hz = (_Float16)0.0f;
  v16h A;
#pragma unroll
  for (int j = 0; j < 16; ++j) A[j] = hz;
  A[0] = lo ? (_Float16)g0 : hz;
  A[1] = lo ? (_Float16)g1 : hz;
  A[2] = lo ? (_Float16)g2 : hz;
  const int n = lane & 15;
  const int rowbase = (lane >> 4) << 3;
#pragma unroll
  for (int nt = 0; nt < NT_H; ++nt) {
    const int col = nt * 16 + n;
    v8f acc = zero_v8f();
    acc = wmma_f16(A, load_btile(W1, nt, lane), acc);
    const float bv = (col < 100) ? b1[col] : 0.0f;
#pragma unroll
    for (int r = 0; r < 8; ++r)
      h[(rowbase + r) * HP + col] = (_Float16)fast_tanh(acc[r] + bv);
  }
}

// Hidden layer: h(16x128) @ W(128x128) + b, tanh, written back in place.
// All 4 A chunks are loaded to registers before any store, so in-place is safe
// (same-wave LDS ops are processed in order).
__device__ __forceinline__ void layer_hidden(_Float16* h,
    const _Float16* __restrict__ W, const float* __restrict__ bias, int lane) {
  v16h A[KT_H];
#pragma unroll
  for (int kt = 0; kt < KT_H; ++kt) A[kt] = load_atile_lds(h, lane, kt);
  const int n = lane & 15;
  const int rowbase = (lane >> 4) << 3;
#pragma unroll
  for (int nt = 0; nt < NT_H; ++nt) {
    const int col = nt * 16 + n;
    v8f acc = zero_v8f();
#pragma unroll
    for (int kt = 0; kt < KT_H; ++kt)
      acc = wmma_f16(A[kt], load_btile(W, kt * NT_H + nt, lane), acc);
    const float bv = (col < 100) ? bias[col] : 0.0f;
#pragma unroll
    for (int r = 0; r < 8; ++r)
      h[(rowbase + r) * HP + col] = (_Float16)fast_tanh(acc[r] + bv);
  }
}

// ----------------------------------------------------------------- phase 2

__global__ __launch_bounds__(128)
void conv_kernel(const float* __restrict__ geo,
                 const int* __restrict__ idx_out,
                 const int* __restrict__ idx_in,
                 int Btiles,
                 const float* __restrict__ feat10,
                 const _Float16* __restrict__ W1, const _Float16* __restrict__ W2,
                 const _Float16* __restrict__ W3, const _Float16* __restrict__ W4,
                 const float* __restrict__ b1, const float* __restrict__ b2,
                 const float* __restrict__ b3, const float* __restrict__ b4,
                 float* __restrict__ out) {
  __shared__ __align__(16) _Float16 hbuf[4][16 * HP];
  __shared__ __align__(16) float    cbuf[4][16 * 4];
  const int lane = threadIdx.x & 31;
  const int wid  = threadIdx.x >> 5;
  _Float16* h = hbuf[wid];
  float*    cf = cbuf[wid];

  const int waveTile = blockIdx.x * 4 + wid;
  const int arow  = waveTile / Btiles;
  const int btile = waveTile % Btiles;
  const int p = lane & 15;

  __builtin_prefetch(W4, 0, 0);

  // ---- per-pair geometry + contraction coefficients (dup on lane & lane+16)
  const int aIdx = idx_out[arow];
  const int bIdx = idx_in[btile * 16 + p];
  const float rx = geo[aIdx * 3 + 0] - geo[bIdx * 3 + 0];
  const float ry = geo[aIdx * 3 + 1] - geo[bIdx * 3 + 1];
  const float rz = geo[aIdx * 3 + 2] - geo[bIdx * 3 + 2];
  const float d  = sqrtf(rx * rx + ry * ry + rz * rz);
  const float safe = (d > 1e-9f) ? d : 1.0f;
  const float mask = (d > 1e-9f) ? 1.0f : 0.0f;
  const float ux = rx / safe, uy = ry / safe, uz = rz / safe;
  const float SQ3 = 1.7320508075688772f, SQ15 = 3.872983346207417f;
  const float SQ5H = 1.118033988749895f;
  const float y1x = SQ3 * ux * mask, y1y = SQ3 * uy * mask, y1z = SQ3 * uz * mask;
  const float y20 = SQ15 * ux * uy * mask;
  const float y21 = SQ15 * uy * uz * mask;
  const float y22 = SQ5H * (3.0f * uz * uz - 1.0f) * mask;
  const float y23 = SQ15 * ux * uz * mask;
  const float y24 = 0.5f * SQ15 * (ux * ux - uy * uy) * mask;
  const float* fr = feat10 + bIdx * 10;
  const float INV = 0.03608439182435161f;  // 1/sqrt(768)
  const float s00 = fr[0] * INV;
  const float s01 = fr[1] * INV;
  const float s1v = (y1x * fr[2] + y1y * fr[3] + y1z * fr[4]) *
                    (INV * 0.5773502691896258f);   // /sqrt(3)
  const float s2v = (y20 * fr[5] + y21 * fr[6] + y22 * fr[7] +
                     y23 * fr[8] + y24 * fr[9]) *
                    (INV * 0.4472135954999579f);   // /sqrt(5)
  if (lane < 16) {
    float* cp = cf + p * 4;
    cp[0] = s00; cp[1] = s01; cp[2] = s1v; cp[3] = s2v;
  }
  asm volatile("s_wait_dscnt 0" ::: "memory");

  // ---- MLP 3 -> 128 -> 128 -> 128 (tanh) via WMMA
  layer_first(h, d, W1, b1, lane);
  layer_hidden(h, W2, b2, lane);
  layer_hidden(h, W3, b3, lane);

  // ---- final layer 128 -> 400 fused with contraction + atomic scatter
  v16h A[KT_H];
#pragma unroll
  for (int kt = 0; kt < KT_H; ++kt) A[kt] = load_atile_lds(h, lane, kt);
  const int n = lane & 15;
  const int rowbase = (lane >> 4) << 3;
  float c0[8], c1[8], c2[8], c3[8];
#pragma unroll
  for (int r = 0; r < 8; ++r) {
    const float* cp = cf + (rowbase + r) * 4;
    c0[r] = cp[0]; c1[r] = cp[1]; c2[r] = cp[2]; c3[r] = cp[3];
  }
  float* outrow = out + (size_t)arow * 100;
#pragma unroll
  for (int nt = 0; nt < 25; ++nt) {
    const int col = nt * 16 + n;
    v8f acc = zero_v8f();
#pragma unroll
    for (int kt = 0; kt < KT_H; ++kt)
      acc = wmma_f16(A[kt], load_btile(W4, kt * 25 + nt, lane), acc);
    const float bv = b4[col];
    float partial = 0.0f;
    int w;
    if (col < 200) {               // r0: Rr[p, 2w+u] * fin[b,u]
      const float* cs = (col & 1) ? c1 : c0;
      w = col >> 1;
#pragma unroll
      for (int r = 0; r < 8; ++r) partial += cs[r] * (acc[r] + bv);
    } else if (col < 300) {        // r1: * (y1 . fin[2:5]) / sqrt3
      w = col - 200;
#pragma unroll
      for (int r = 0; r < 8; ++r) partial += c2[r] * (acc[r] + bv);
    } else {                       // r2: * (y2 . fin[5:10]) / sqrt5
      w = col - 300;
#pragma unroll
      for (int r = 0; r < 8; ++r) partial += c3[r] * (acc[r] + bv);
    }
    atomicAdd(outrow + w, partial);
  }
}

// ----------------------------------------------------------------- phase 1

__global__ __launch_bounds__(128)
void phase1_kernel(const float* __restrict__ geo,
                   const float* __restrict__ feat,
                   const _Float16* __restrict__ W1, const _Float16* __restrict__ W2,
                   const _Float16* __restrict__ W3, const _Float16* __restrict__ W4,
                   const float* __restrict__ b1, const float* __restrict__ b2,
                   const float* __restrict__ b3, const float* __restrict__ b4,
                   float* __restrict__ xbuf) {
  __shared__ __align__(16) _Float16 hbuf[4][16 * HP];
  __shared__ __align__(16) float    rbuf[4][16 * 24];
  const int lane = threadIdx.x & 31;
  const int wid  = threadIdx.x >> 5;
  _Float16* h = hbuf[wid];
  float*    Rt = rbuf[wid];

  const int Btiles = 48;  // 768/16
  const int waveTile = blockIdx.x * 4 + wid;
  const int a = waveTile / Btiles;
  const int btile = waveTile % Btiles;
  const int p = lane & 15;
  const int b = btile * 16 + p;

  const float rx = geo[a * 3 + 0] - geo[b * 3 + 0];
  const float ry = geo[a * 3 + 1] - geo[b * 3 + 1];
  const float rz = geo[a * 3 + 2] - geo[b * 3 + 2];
  const float d  = sqrtf(rx * rx + ry * ry + rz * rz);
  const float safe = (d > 1e-9f) ? d : 1.0f;
  const float mask = (d > 1e-9f) ? 1.0f : 0.0f;
  const float ux = rx / safe, uy = ry / safe, uz = rz / safe;
  const float SQ3 = 1.7320508075688772f, SQ15 = 3.872983346207417f;
  const float SQ5H = 1.118033988749895f;
  const float y1x = SQ3 * ux * mask, y1y = SQ3 * uy * mask, y1z = SQ3 * uz * mask;
  const float y20 = SQ15 * ux * uy * mask;
  const float y21 = SQ15 * uy * uz * mask;
  const float y22 = SQ5H * (3.0f * uz * uz - 1.0f) * mask;
  const float y23 = SQ15 * ux * uz * mask;
  const float y24 = 0.5f * SQ15 * (ux * ux - uy * uy) * mask;
  const float fb0 = feat[b * 4 + 0], fb1 = feat[b * 4 + 1];
  const float fb2 = feat[b * 4 + 2], fb3 = feat[b * 4 + 3];

  layer_first(h, d, W1, b1, lane);
  layer_hidden(h, W2, b2, lane);
  layer_hidden(h, W3, b3, lane);

  // final 128 -> 24 (pad 32): store R rows to LDS
  v16h A[KT_H];
#pragma unroll
  for (int kt = 0; kt < KT_H; ++kt) A[kt] = load_atile_lds(h, lane, kt);
  const int n = lane & 15;
  const int rowbase = (lane >> 4) << 3;
#pragma unroll
  for (int nt = 0; nt < 2; ++nt) {
    const int col = nt * 16 + n;
    v8f acc = zero_v8f();
#pragma unroll
    for (int kt = 0; kt < KT_H; ++kt)
      acc = wmma_f16(A[kt], load_btile(W4, kt * 2 + nt, lane), acc);
    if (col < 24) {
      const float bv = b4[col];
#pragma unroll
      for (int r = 0; r < 8; ++r)
        Rt[(rowbase + r) * 24 + col] = acc[r] + bv;
    }
  }
  asm volatile("s_wait_dscnt 0" ::: "memory");

  if (lane < 16) {  // one lane per pair: tiny contractions into x[a, 0..11]
    const float* R = Rt + p * 24;
    float* xp = xbuf + (size_t)a * 12;
#pragma unroll
    for (int u = 0; u < 4; ++u) {
      const float c = R[4 * u + 0] * fb0 + R[4 * u + 1] * fb1 +
                      R[4 * u + 2] * fb2 + R[4 * u + 3] * fb3;
      atomicAdd(xp + u, c);
    }
    const float s1 = R[16] * fb0 + R[17] * fb1 + R[18] * fb2 + R[19] * fb3;
    atomicAdd(xp + 4, s1 * y1x);
    atomicAdd(xp + 5, s1 * y1y);
    atomicAdd(xp + 6, s1 * y1z);
    const float s2 = R[20] * fb0 + R[21] * fb1 + R[22] * fb2 + R[23] * fb3;
    atomicAdd(xp + 7,  s2 * y20);
    atomicAdd(xp + 8,  s2 * y21);
    atomicAdd(xp + 9,  s2 * y22);
    atomicAdd(xp + 10, s2 * y23);
    atomicAdd(xp + 11, s2 * y24);
  }
}

// -------------------------------------------------- gating: x -> feat10
// (once over 768 rows; keep precise libm forms here)

__global__ void act_kernel(const float* __restrict__ x, float* __restrict__ f10) {
  const int i = blockIdx.x * blockDim.x + threadIdx.x;
  if (i >= 768) return;
  const float INV = 0.03608439182435161f;
  const float* xr = x + (size_t)i * 12;
  float v[12];
#pragma unroll
  for (int j = 0; j < 12; ++j) v[j] = xr[j] * INV;
  float* o = f10 + (size_t)i * 10;
  o[0] = tanhf(v[0]);
  o[1] = tanhf(v[1]);
  const float g0 = 1.0f / (1.0f + expf(-v[2]));
  const float g1 = 1.0f / (1.0f + expf(-v[3]));
  o[2] = v[4] * g0; o[3] = v[5] * g0; o[4] = v[6] * g0;
  o[5] = v[7] * g1; o[6] = v[8] * g1; o[7] = v[9] * g1;
  o[8] = v[10] * g1; o[9] = v[11] * g1;
}

// -------------------------------------------------- weight packing (f32->f16)
// dst element idx: j = idx&15 (element in lane), lane = (idx>>4)&31,
// tile = idx>>9 with tile = kt*Ntiles + nt.
// B layout: lane column = lane&15; K = kt*32 + (lane>=16 ? 8 : 0) + (j<8 ? j : j+8).

__global__ void pack_kernel(const float* __restrict__ W, int Kin, int N,
                            _Float16* __restrict__ dst, int Ntiles, int total) {
  const int idx = blockIdx.x * 256 + threadIdx.x;
  if (idx >= total) return;
  const int j    = idx & 15;
  const int lane = (idx >> 4) & 31;
  const int tile = idx >> 9;
  const int nt = tile % Ntiles;
  const int kt = tile / Ntiles;
  const int col = nt * 16 + (lane & 15);
  const int k   = kt * 32 + ((lane >= 16) ? 8 : 0) + ((j < 8) ? j : j + 8);
  const float v = (k < Kin && col < N) ? W[(size_t)k * N + col] : 0.0f;
  dst[idx] = (_Float16)v;
}

// ----------------------------------------------------------------- launcher

extern "C" void kernel_launch(void* const* d_in, const int* in_sizes, int n_in,
                              void* d_out, int out_size, void* d_ws, size_t ws_size,
                              hipStream_t stream) {
  const float* feat = (const float*)d_in[0];
  const float* geo  = (const float*)d_in[1];
  const int*   idxO = (const int*)d_in[2];
  const int*   idxH = (const int*)d_in[3];
  // params flattened in setup_inputs insertion order:
  // mlp order: l1, m0_O, m0_H, m1_O, m1_H; per layer: W then b (4 layers each)
  const float* P[40];
  for (int i = 0; i < 40; ++i)
    P[i] = (4 + i < n_in) ? (const float*)d_in[4 + i] : nullptr;

  char* ws = (char*)d_ws;
  size_t off = 0;
  _Float16* Wp[5][4];
  const int w4nt[5] = {2, 25, 25, 25, 25};  // N-tiles of last layer (32 / 400)
  for (int m = 0; m < 5; ++m) {
    Wp[m][0] = (_Float16*)(ws + off); off += (size_t)1 * 8 * 512 * 2;          // 3->128
    Wp[m][1] = (_Float16*)(ws + off); off += (size_t)4 * 8 * 512 * 2;          // 128->128
    Wp[m][2] = (_Float16*)(ws + off); off += (size_t)4 * 8 * 512 * 2;          // 128->128
    Wp[m][3] = (_Float16*)(ws + off); off += (size_t)4 * w4nt[m] * 512 * 2;    // 128->out
  }
  float* xbuf = (float*)(ws + off); off += (size_t)768 * 12 * 4;
  float* f10  = (float*)(ws + off); off += (size_t)768 * 10 * 4;
  (void)ws_size; (void)in_sizes;

  // pack all weights to WMMA B layout (runs every call: deterministic)
  for (int m = 0; m < 5; ++m) {
    const int nout4 = (m == 0) ? 24 : 400;
    const int t1 = 1 * 8 * 512;
    const int t2 = 4 * 8 * 512;
    const int t4 = 4 * w4nt[m] * 512;
    pack_kernel<<<(t1 + 255) / 256, 256, 0, stream>>>(P[m * 8 + 0], 3, 100,
                                                      Wp[m][0], 8, t1);
    pack_kernel<<<(t2 + 255) / 256, 256, 0, stream>>>(P[m * 8 + 2], 100, 100,
                                                      Wp[m][1], 8, t2);
    pack_kernel<<<(t2 + 255) / 256, 256, 0, stream>>>(P[m * 8 + 4], 100, 100,
                                                      Wp[m][2], 8, t2);
    pack_kernel<<<(t4 + 255) / 256, 256, 0, stream>>>(P[m * 8 + 6], 100, nout4,
                                                      Wp[m][3], w4nt[m], t4);
  }

  hipMemsetAsync(xbuf, 0, (size_t)768 * 12 * 4, stream);
  hipMemsetAsync(d_out, 0, (size_t)out_size * sizeof(float), stream);

  // phase 1: 768x768 pairs, 16 pairs/wave, 4 waves/block
  phase1_kernel<<<9216, 128, 0, stream>>>(geo, feat,
      Wp[0][0], Wp[0][1], Wp[0][2], Wp[0][3],
      P[1], P[3], P[5], P[7], xbuf);
  act_kernel<<<3, 256, 0, stream>>>(xbuf, f10);

  float* outO = (float*)d_out;
  float* outH = outO + (size_t)256 * 100;

  // m0_O: gout=O(256), gin=O(16 tiles)
  conv_kernel<<<1024, 128, 0, stream>>>(geo, idxO, idxO, 16, f10,
      Wp[1][0], Wp[1][1], Wp[1][2], Wp[1][3],
      P[9], P[11], P[13], P[15], outO);
  // m0_H: gout=O(256), gin=H(32 tiles)
  conv_kernel<<<2048, 128, 0, stream>>>(geo, idxO, idxH, 32, f10,
      Wp[2][0], Wp[2][1], Wp[2][2], Wp[2][3],
      P[17], P[19], P[21], P[23], outO);
  // m1_O: gout=H(512), gin=O(16 tiles)
  conv_kernel<<<2048, 128, 0, stream>>>(geo, idxH, idxO, 16, f10,
      Wp[3][0], Wp[3][1], Wp[3][2], Wp[3][3],
      P[25], P[27], P[29], P[31], outH);
  // m1_H: gout=H(512), gin=H(32 tiles)
  conv_kernel<<<4096, 128, 0, stream>>>(geo, idxH, idxH, 32, f10,
      Wp[4][0], Wp[4][1], Wp[4][2], Wp[4][3],
      P[33], P[35], P[37], P[39], outH);
}
